// RGLRU_28157805592826
// MI455X (gfx1250) — matrix operand order repaired and checked by reading, not compiled, not run on hardware
//
#include <hip/hip_runtime.h>
#include <hip/hip_bf16.h>
#include <stdint.h>

// ---------------------------------------------------------------------------
// RG-LRU forward for MI455X (gfx1250, wave32, WMMA + async global->LDS)
//   B=2, S=4096, D=2048.  Gate GEMMs (8192x2048x2048, x2) on v_wmma bf16.
// ---------------------------------------------------------------------------

typedef __attribute__((ext_vector_type(16))) __bf16 v16bf;
typedef __attribute__((ext_vector_type(8)))  __bf16 v8bf;
typedef __attribute__((ext_vector_type(8)))  float  v8f;
typedef int v4i __attribute__((vector_size(16)));   // matches builtin param type

constexpr int   BSZ = 2, SEQLEN = 4096, DIM = 2048, CONVK = 4, HDIM = 128;
constexpr int   Mrows = BSZ * SEQLEN;           // 8192 GEMM rows
constexpr float CCONST = 8.0f;

// GEMM tiling
constexpr int BM = 128, BN = 64, BK = 32;
constexpr int NK = DIM / BK;                    // 64 K-steps
constexpr int LDT = 40;                         // LDS pitch in halves (80B, conflict-free)

// Scan chunking
constexpr int CL = 128, NC = SEQLEN / CL;       // 32 chunks of 128 steps

// Workspace layout (bytes)
constexpr size_t NELT    = (size_t)BSZ * SEQLEN * DIM;          // 16,777,216
constexpr size_t OFF_XCF = 0;                                   // conv out f32
constexpr size_t OFF_XCB = OFF_XCF + NELT * 4;                  // conv out bf16
constexpr size_t OFF_WIB = OFF_XCB + NELT * 2;                  // W_ig bf16
constexpr size_t OFF_WAB = OFF_WIB + (size_t)DIM * DIM * 2;     // W_ag bf16
constexpr size_t OFF_AB  = OFF_WAB + (size_t)DIM * DIM * 2;     // (a_eff, b) float2
constexpr size_t OFF_CHK = OFF_AB + NELT * 8;                   // chunk (A,B) float2
constexpr size_t OFF_CAR = OFF_CHK + (size_t)BSZ * NC * DIM * 8;// chunk carries f32
constexpr size_t OFF_MSK = OFF_CAR + (size_t)BSZ * NC * DIM * 4;// reset mask f32

#if defined(__gfx1250__) && __has_builtin(__builtin_amdgcn_global_load_async_to_lds_b128)
#define USE_ASYNC_LDS 1
#else
#define USE_ASYNC_LDS 0
#endif

static __device__ __forceinline__ unsigned short f2bf(float f) {
    unsigned u = __builtin_bit_cast(unsigned, f);
    u += 0x7fffu + ((u >> 16) & 1u);            // round-to-nearest-even
    return (unsigned short)(u >> 16);
}
static __device__ __forceinline__ float sigmoidf_fast(float x) {
    return 1.0f / (1.0f + __expf(-x));
}
static __device__ __forceinline__ float softplusf(float x) {
    return __logf(1.0f + __expf(x));
}

#if USE_ASYNC_LDS
static __device__ __forceinline__ void cp_async16(unsigned short* lds,
                                                  const unsigned short* g) {
    // global_load_async_to_lds_b128: 16B/lane, LDS-direct, tracked by ASYNCcnt
    __builtin_amdgcn_global_load_async_to_lds_b128(
        (__attribute__((address_space(1))) v4i*)(g),
        (__attribute__((address_space(3))) v4i*)(lds), 0, 0);
}
static __device__ __forceinline__ void wait_async0() {
#if __has_builtin(__builtin_amdgcn_s_wait_asynccnt)
    __builtin_amdgcn_s_wait_asynccnt(0);
#else
    asm volatile("s_wait_asynccnt 0x0" ::: "memory");
#endif
}
#endif

// ---------------------------------------------------------------------------
// 1) depthwise causal conv (K=4) + bf16 pack of the result
// ---------------------------------------------------------------------------
__global__ void k_conv(const float* __restrict__ x, const float* __restrict__ w,
                       float* __restrict__ xcf, unsigned short* __restrict__ xcb) {
    size_t idx = (size_t)blockIdx.x * blockDim.x + threadIdx.x;
    int d  = (int)(idx % DIM);
    int st = (int)(idx / DIM);
    int t  = st % SEQLEN;
    int b  = st / SEQLEN;
    float acc = 0.0f;
#pragma unroll
    for (int k = 0; k < CONVK; ++k) {
        int tt = t - (CONVK - 1) + k;
        if (tt >= 0)
            acc = fmaf(w[d * CONVK + k], x[((size_t)b * SEQLEN + tt) * DIM + d], acc);
    }
    xcf[idx] = acc;
    xcb[idx] = f2bf(acc);
}

// ---------------------------------------------------------------------------
// 2) weight f32 -> bf16 convert
// ---------------------------------------------------------------------------
__global__ void k_wcvt(const float* __restrict__ src, unsigned short* __restrict__ dst) {
    size_t idx = (size_t)blockIdx.x * blockDim.x + threadIdx.x;
    dst[idx] = f2bf(src[idx]);
}

// ---------------------------------------------------------------------------
// 3) reset mask from cu_seqlens (reset at each packed-sequence start)
// ---------------------------------------------------------------------------
__global__ void k_mask(const int* __restrict__ cu, int ncu, float* __restrict__ msk) {
    int s = blockIdx.x * blockDim.x + threadIdx.x;
    float m = 0.0f;
    for (int i = 0; i < ncu - 1; ++i)
        if (cu[i] == s) m = 1.0f;
    msk[s] = m;
}

// ---------------------------------------------------------------------------
// 4) dual gate GEMM (bf16 WMMA, f32 accum) + fused RG-LRU elementwise epilogue
//    block = 256 threads = 8 waves; tile = 128(M) x 64(N) for BOTH gates.
//    Double-buffered LDS; stage-in via async global->LDS when available.
// ---------------------------------------------------------------------------
__global__ __launch_bounds__(256) void k_gates(
    const unsigned short* __restrict__ xcb,   // [M, K] bf16 conv output
    const unsigned short* __restrict__ wib,   // [E, K] bf16
    const unsigned short* __restrict__ wab,   // [E, K] bf16
    const float* __restrict__ xcf,            // [M, K] f32 conv output
    const float* __restrict__ a_param,        // [HDIM]
    const float* __restrict__ msk,            // [S]
    float2* __restrict__ ab)                  // [M, E] (a_eff, b)
{
    __shared__ unsigned short sA [2][BM * LDT];  // 2 x 10.0 KB
    __shared__ unsigned short sBi[2][BN * LDT];  // 2 x  5.0 KB
    __shared__ unsigned short sBa[2][BN * LDT];  // 2 x  5.0 KB

    const int tid  = threadIdx.x;
    const int lane = tid & 31, wave = tid >> 5;
    const int lm   = lane & 15, lh = lane >> 4;
    const int tm   = blockIdx.x % (Mrows / BM);
    const int tn   = blockIdx.x / (Mrows / BM);
    const int m0   = tm * BM, n0 = tn * BN;

    v8f accI[4], accA[4];
#pragma unroll
    for (int nt = 0; nt < 4; ++nt) {
        accI[nt] = (v8f){0.f,0.f,0.f,0.f,0.f,0.f,0.f,0.f};
        accA[nt] = (v8f){0.f,0.f,0.f,0.f,0.f,0.f,0.f,0.f};
    }

    const int arow  = tid >> 2;                // 0..63
    const int apart = tid & 3;                 // 16B chunk within a 64B row slab
    const int lo    = arow * LDT + apart * 8;  // LDS dst offset (halves)

    const unsigned short* gA0 = xcb + (size_t)(m0 + arow)      * DIM + apart * 8;
    const unsigned short* gA1 = xcb + (size_t)(m0 + 64 + arow) * DIM + apart * 8;
    const unsigned short* gBi = wib + (size_t)(n0 + arow)      * DIM + apart * 8;
    const unsigned short* gBa = wab + (size_t)(n0 + arow)      * DIM + apart * 8;

    union Frag { v8bf h[2]; v16bf v; };

    // ---- prologue: stage K-slab 0 into buffer 0 ----
#if USE_ASYNC_LDS
    cp_async16(&sA [0][lo],            gA0);
    cp_async16(&sA [0][64 * LDT + lo], gA1);
    cp_async16(&sBi[0][lo],            gBi);
    cp_async16(&sBa[0][lo],            gBa);
    wait_async0();
#else
    {
        uint4 ra0 = *(const uint4*)gA0;
        uint4 ra1 = *(const uint4*)gA1;
        uint4 rbi = *(const uint4*)gBi;
        uint4 rba = *(const uint4*)gBa;
        *(uint4*)&sA [0][lo]            = ra0;
        *(uint4*)&sA [0][64 * LDT + lo] = ra1;
        *(uint4*)&sBi[0][lo]            = rbi;
        *(uint4*)&sBa[0][lo]            = rba;
    }
#endif
    __syncthreads();

    for (int kk = 0; kk < NK; ++kk) {
        const int  cur  = kk & 1;
        const int  nxt  = cur ^ 1;
        const bool more = (kk + 1 < NK);
        const int  k1   = (kk + 1) * BK;

        // ---- stage next K-slab into the other buffer while computing ----
#if USE_ASYNC_LDS
        if (more) {
            cp_async16(&sA [nxt][lo],            gA0 + k1);
            cp_async16(&sA [nxt][64 * LDT + lo], gA1 + k1);
            cp_async16(&sBi[nxt][lo],            gBi + k1);
            cp_async16(&sBa[nxt][lo],            gBa + k1);
        }
#else
        uint4 ra0, ra1, rbi, rba;
        if (more) {
            ra0 = *(const uint4*)(gA0 + k1);
            ra1 = *(const uint4*)(gA1 + k1);
            rbi = *(const uint4*)(gBi + k1);
            rba = *(const uint4*)(gBa + k1);
        }
#endif

        // ---- WMMA compute on current buffer ----
        // A fragment (16x32 bf16): lane lh gets K in [16h, 16h+16)
        const unsigned short* aptr = &sA[cur][(size_t)(wave * 16 + lm) * LDT];
        Frag af;
        af.h[0] = *(const v8bf*)(aptr + lh * 16);
        af.h[1] = *(const v8bf*)(aptr + lh * 16 + 8);

#pragma unroll
        for (int nt = 0; nt < 4; ++nt) {
            // B fragment (32x16 bf16): lane (n=lm, h=lh) gets K in [16h, 16h+16)
            const unsigned short* bpi = &sBi[cur][(size_t)(nt * 16 + lm) * LDT + lh * 16];
            Frag bi;
            bi.h[0] = *(const v8bf*)(bpi);
            bi.h[1] = *(const v8bf*)(bpi + 8);
            accI[nt] = __builtin_amdgcn_wmma_f32_16x16x32_bf16(
                false, af.v, false, bi.v, (short)0, accI[nt], false, false);

            const unsigned short* bpa = &sBa[cur][(size_t)(nt * 16 + lm) * LDT + lh * 16];
            Frag ba;
            ba.h[0] = *(const v8bf*)(bpa);
            ba.h[1] = *(const v8bf*)(bpa + 8);
            accA[nt] = __builtin_amdgcn_wmma_f32_16x16x32_bf16(
                false, af.v, false, ba.v, (short)0, accA[nt], false, false);
        }

        // ---- publish next buffer ----
#if USE_ASYNC_LDS
        if (more) wait_async0();
#else
        if (more) {
            *(uint4*)&sA [nxt][lo]            = ra0;
            *(uint4*)&sA [nxt][64 * LDT + lo] = ra1;
            *(uint4*)&sBi[nxt][lo]            = rbi;
            *(uint4*)&sBa[nxt][lo]            = rba;
        }
#endif
        __syncthreads();
    }

    // ---- fused RG-LRU elementwise epilogue ----
#pragma unroll
    for (int nt = 0; nt < 4; ++nt) {
        const int col = n0 + nt * 16 + lm;
        const float afull = softplusf(a_param[col & (HDIM - 1)]);
#pragma unroll
        for (int r = 0; r < 8; ++r) {
            const int row = m0 + wave * 16 + 8 * lh + r;   // C/D layout: M = r + 8*(lane/16)
            float gx   = sigmoidf_fast(accI[nt][r]);
            float ga   = sigmoidf_fast(accA[nt][r]);
            float la   = -CCONST * ga * afull;             // log_a
            float av   = __expf(la);
            float mult = sqrtf(fmaxf(1.0f - __expf(2.0f * la), 0.0f));
            float xv   = xcf[(size_t)row * DIM + col];
            float bv   = mult * gx * xv;
            const int t = row & (SEQLEN - 1);
            float aeff = av * (1.0f - msk[t]);
            ab[(size_t)row * DIM + col] = make_float2(aeff, bv);
        }
    }
}

// ---------------------------------------------------------------------------
// 5) scan pass 1: per-(b, chunk, d) reduce -> (A = prod a, B = local scan tail)
// ---------------------------------------------------------------------------
__global__ void k_chunk(const float2* __restrict__ ab, float2* __restrict__ chk) {
    int idx = blockIdx.x * blockDim.x + threadIdx.x;    // B*NC*DIM
    int d = idx % DIM;
    int c = (idx / DIM) % NC;
    int b = idx / (DIM * NC);
    size_t base = ((size_t)b * SEQLEN + (size_t)c * CL) * DIM + d;
    float A = 1.0f, Bv = 0.0f;
#pragma unroll 4
    for (int t = 0; t < CL; ++t) {
        float2 v = ab[base + (size_t)t * DIM];
        A *= v.x;
        Bv = fmaf(v.x, Bv, v.y);
    }
    chk[((size_t)b * NC + c) * DIM + d] = make_float2(A, Bv);
}

// ---------------------------------------------------------------------------
// 6) scan pass 2: serial carry scan over the 32 chunks (tiny)
// ---------------------------------------------------------------------------
__global__ void k_carry(const float2* __restrict__ chk, float* __restrict__ car) {
    int idx = blockIdx.x * blockDim.x + threadIdx.x;    // B*DIM
    int d = idx % DIM;
    int b = idx / DIM;
    float h = 0.0f;
    for (int c = 0; c < NC; ++c) {
        size_t o = ((size_t)b * NC + c) * DIM + d;
        car[o] = h;                                     // state entering chunk c
        float2 v = chk[o];
        h = fmaf(v.x, h, v.y);
    }
}

// ---------------------------------------------------------------------------
// 7) scan pass 3: apply recurrence within each chunk starting from its carry
// ---------------------------------------------------------------------------
__global__ void k_apply(const float2* __restrict__ ab, const float* __restrict__ car,
                        float* __restrict__ out) {
    int idx = blockIdx.x * blockDim.x + threadIdx.x;    // B*NC*DIM
    int d = idx % DIM;
    int c = (idx / DIM) % NC;
    int b = idx / (DIM * NC);
    size_t base = ((size_t)b * SEQLEN + (size_t)c * CL) * DIM + d;
    float h = car[((size_t)b * NC + c) * DIM + d];
#pragma unroll 4
    for (int t = 0; t < CL; ++t) {
        float2 v = ab[base + (size_t)t * DIM];
        h = fmaf(v.x, h, v.y);
        out[base + (size_t)t * DIM] = h;
    }
}

// ---------------------------------------------------------------------------
extern "C" void kernel_launch(void* const* d_in, const int* in_sizes, int n_in,
                              void* d_out, int out_size, void* d_ws, size_t ws_size,
                              hipStream_t stream) {
    const float* x   = (const float*)d_in[0];
    const float* cw  = (const float*)d_in[1];
    const float* wig = (const float*)d_in[2];
    const float* wag = (const float*)d_in[3];
    const float* ap  = (const float*)d_in[4];
    const int*   cu  = (const int*)d_in[5];
    const int    ncu = in_sizes[5];

    char* ws = (char*)d_ws;
    float*          xcf = (float*)         (ws + OFF_XCF);
    unsigned short* xcb = (unsigned short*)(ws + OFF_XCB);
    unsigned short* wib = (unsigned short*)(ws + OFF_WIB);
    unsigned short* wab = (unsigned short*)(ws + OFF_WAB);
    float2*         abv = (float2*)        (ws + OFF_AB);
    float2*         chk = (float2*)        (ws + OFF_CHK);
    float*          car = (float*)         (ws + OFF_CAR);
    float*          msk = (float*)         (ws + OFF_MSK);
    float*          out = (float*)d_out;

    k_conv <<<(int)(NELT / 256), 256, 0, stream>>>(x, cw, xcf, xcb);
    k_wcvt <<<DIM * DIM / 256, 256, 0, stream>>>(wig, wib);
    k_wcvt <<<DIM * DIM / 256, 256, 0, stream>>>(wag, wab);
    k_mask <<<SEQLEN / 256, 256, 0, stream>>>(cu, ncu, msk);
    k_gates<<<(Mrows / BM) * (DIM / BN), 256, 0, stream>>>(xcb, wib, wab, xcf, ap, msk, abv);
    k_chunk<<<BSZ * NC * DIM / 256, 256, 0, stream>>>(abv, chk);
    k_carry<<<BSZ * DIM / 256, 256, 0, stream>>>(chk, car);
    k_apply<<<BSZ * NC * DIM / 256, 256, 0, stream>>>(abv, car, out);
}
